// DCANet_1795296329879
// MI455X (gfx1250) — matrix operand order repaired and checked by reading, not compiled, not run on hardware
//
#include <hip/hip_runtime.h>
#include <hip/hip_bf16.h>
#include <math.h>

// ---------------------------------------------------------------------------
// Problem constants
// ---------------------------------------------------------------------------
#define WAY 5
#define NQ  75
#define CCH 640
#define HW  100          // H*W = 10*10
#define SEH 40           // C / SE_R
#define ROWS 500         // WAY*HW
#define ROWS_PAD 512     // padded to 32 row-tiles of 16
#define COLS_PAD 112     // HW padded to 7 col-tiles of 16

// ---------------------------------------------------------------------------
// Workspace layout (bytes)
// ---------------------------------------------------------------------------
static constexpr size_t ALGN(size_t x) { return (x + 255) & ~(size_t)255; }
static constexpr size_t SPT_SE_OFF = 0;                                               // [5][640][100] f32
static constexpr size_t QRY_SE_OFF = SPT_SE_OFF + ALGN((size_t)WAY*CCH*HW*4);         // [75][640][100] f32
static constexpr size_t CORR_OFF   = QRY_SE_OFF + ALGN((size_t)NQ*CCH*HW*4);          // [75][5][100][100] f32
static constexpr size_t ATTN_S_OFF = CORR_OFF   + ALGN((size_t)NQ*WAY*HW*HW*4);       // [375][100] f32
static constexpr size_t ATTN_Q_OFF = ATTN_S_OFF + ALGN((size_t)NQ*WAY*HW*4);          // [375][100] f32
static constexpr size_t SN16_OFF   = ATTN_Q_OFF + ALGN((size_t)NQ*WAY*HW*4);          // [512][640] f16
static constexpr size_t QN16_OFF   = SN16_OFF   + (size_t)ROWS_PAD*CCH*2;             // [75][112][640] f16
static constexpr size_t F16_BYTES  = (size_t)ROWS_PAD*CCH*2 + (size_t)NQ*COLS_PAD*CCH*2;

// ---------------------------------------------------------------------------
// WMMA vector types (gfx1250, wave32)
// ---------------------------------------------------------------------------
typedef __attribute__((ext_vector_type(8)))  _Float16 h8;
typedef __attribute__((ext_vector_type(16))) _Float16 v16h;
typedef __attribute__((ext_vector_type(8)))  float    v8f;

// ---------------------------------------------------------------------------
// K0: zero the f16 buffers (pads must be zero so WMMA tiles load unguarded)
// ---------------------------------------------------------------------------
__global__ void k0_zero(unsigned int* __restrict__ p, size_t nwords) {
  size_t i = (size_t)blockIdx.x * blockDim.x + threadIdx.x;
  size_t st = (size_t)gridDim.x * blockDim.x;
  for (; i < nwords; i += st) p[i] = 0u;
}

// ---------------------------------------------------------------------------
// K1: per-sample preprocess:
//   channel-mean subtract -> SE (pool/fc/relu/fc/sigmoid/scale) ->
//   f32 SE output [c][p]  +  f16 L2-normalized, position-major [p][c]
// blocks 0..4 = support samples, 5..79 = query samples
// ---------------------------------------------------------------------------
__global__ void k1_pre(const float* __restrict__ spt, const float* __restrict__ qry,
                       const float* __restrict__ w1,  const float* __restrict__ b1,
                       const float* __restrict__ w2,  const float* __restrict__ b2,
                       float* __restrict__ spt_se, float* __restrict__ qry_se,
                       _Float16* __restrict__ sn16, _Float16* __restrict__ qn16) {
  __shared__ float m[HW];      // per-position channel mean
  __shared__ float sq[CCH];    // squeeze vector
  __shared__ float hh[SEH];    // hidden
  __shared__ float gg[CCH];    // gate
  __shared__ float inrm[HW];   // position L2 norm^2 -> 1/norm
  __shared__ float mbar;

  const int b    = blockIdx.x;
  const bool isS = (b < WAY);
  const int sidx = isS ? b : (b - WAY);
  const float* X = (isS ? spt : qry) + (size_t)sidx * CCH * HW;
  float*       Y = (isS ? spt_se : qry_se) + (size_t)sidx * CCH * HW;
  _Float16*    O = isS ? (sn16 + (size_t)sidx * HW * CCH)
                       : (qn16 + (size_t)sidx * COLS_PAD * CCH);
  const int tid = threadIdx.x;
  const int nt  = blockDim.x;

  // per-position mean over channels
  for (int p = tid; p < HW; p += nt) {
    float s = 0.f;
    for (int c = 0; c < CCH; ++c) s += X[c * HW + p];
    m[p] = s * (1.0f / CCH);
  }
  __syncthreads();
  if (tid == 0) { float t = 0.f; for (int p = 0; p < HW; ++p) t += m[p]; mbar = t * (1.0f / HW); }
  __syncthreads();

  // SE squeeze: global average of (x - m)
  for (int c = tid; c < CCH; c += nt) {
    float s = 0.f; const float* xr = X + c * HW;
    for (int p = 0; p < HW; ++p) s += xr[p];
    sq[c] = s * (1.0f / HW) - mbar;
  }
  __syncthreads();

  // fc1 + relu
  for (int j = tid; j < SEH; j += nt) {
    float s = b1[j]; const float* wr = w1 + j * CCH;
    for (int c = 0; c < CCH; ++c) s += wr[c] * sq[c];
    hh[j] = fmaxf(s, 0.f);
  }
  __syncthreads();

  // fc2 + sigmoid
  for (int c = tid; c < CCH; c += nt) {
    float s = b2[c]; const float* wr = w2 + c * SEH;
    for (int j = 0; j < SEH; ++j) s += wr[j] * hh[j];
    gg[c] = 1.0f / (1.0f + __expf(-s));
  }
  for (int p = tid; p < HW; p += nt) inrm[p] = 0.f;
  __syncthreads();

  // SE output + per-position squared norms (LDS float atomics -> ds_add_f32)
  for (int idx = tid; idx < CCH * HW; idx += nt) {
    int c = idx / HW, p = idx - c * HW;
    float y = (X[idx] - m[p]) * gg[c];
    Y[idx] = y;
    atomicAdd(&inrm[p], y * y);
  }
  __syncthreads();
  for (int p = tid; p < HW; p += nt)
    inrm[p] = 1.0f / fmaxf(sqrtf(inrm[p]), 1e-8f);
  __syncthreads();

  // normalized f16, position-major [p][c] (coalesced stores)
  for (int idx = tid; idx < HW * CCH; idx += nt) {
    int p = idx / CCH, c = idx - p * CCH;
    float y = (X[c * HW + p] - m[p]) * gg[c];
    O[(size_t)p * CCH + c] = (_Float16)(y * inrm[p]);
  }
}

// ---------------------------------------------------------------------------
// K2: corr[q,n,ij,kl] = sum_c sn[n,c,ij]*qn[q,c,kl]  via v_wmma_f32_16x16x32_f16
// One wave per 16x16 output tile. A = sn16 [512][640], B = qn16[q] [112][640].
// ---------------------------------------------------------------------------
__global__ void k2_corr_wmma(const _Float16* __restrict__ sn16,
                             const _Float16* __restrict__ qn16,
                             float* __restrict__ corr) {
  const int lane    = threadIdx.x;        // 0..31 (wave32)
  const int colBase = blockIdx.x * 16;    // query-position tile
  const int rowBase = blockIdx.y * 16;    // support-row tile
  const int q       = blockIdx.z;

  const int l15  = lane & 15;
  const int aK   = (lane < 16) ? 0 : 16;  // A frag: K 0-15 vs 16-31
  const int bK   = (lane < 16) ? 0 : 16;  // B frag: K 0-15 vs 16-31
  const _Float16* Abase = sn16 + (size_t)(rowBase + l15) * CCH;
  const _Float16* Bbase = qn16 + ((size_t)q * COLS_PAD + colBase + l15) * CCH;

  v8f acc = {};
#pragma unroll
  for (int kt = 0; kt < CCH / 32; ++kt) {
    const int k0 = kt * 32;
    h8 a0 = *(const h8*)(Abase + k0 + aK);
    h8 a1 = *(const h8*)(Abase + k0 + aK + 8);
    h8 b0 = *(const h8*)(Bbase + k0 + bK);
    h8 b1 = *(const h8*)(Bbase + k0 + bK + 8);
    v16h a, bb;
#pragma unroll
    for (int i = 0; i < 8; ++i) {
      a[i] = a0[i]; a[i + 8] = a1[i];
      bb[i] = b0[i]; bb[i + 8] = b1[i];
    }
    acc = __builtin_amdgcn_wmma_f32_16x16x32_f16(
        /*neg_a=*/false, a, /*neg_b=*/false, bb,
        /*c_mod=*/(short)0, acc, /*reuse_a=*/false, /*reuse_b=*/false);
  }

  // C/D layout: lane<16 -> rows M=0..7, lane>=16 -> rows M=8..15; N = lane&15
  const int col  = colBase + l15;
  const int rOff = (lane < 16) ? 0 : 8;
  if (col < HW) {
#pragma unroll
    for (int i = 0; i < 8; ++i) {
      int r = rowBase + rOff + i;
      if (r < ROWS) {
        int n = r / HW, ij = r - n * HW;
        corr[(((size_t)q * WAY + n) * HW + ij) * HW + col] = acc[i];
      }
    }
  }
}

// ---------------------------------------------------------------------------
// K3: per (q,n): gaussian-normalize (ddof=1) /5 -> softmax over ij (cols) and
// over kl (rows) of the 100x100 corr tile held in LDS; sum out the other axis.
// ---------------------------------------------------------------------------
__global__ void k3_attn(const float* __restrict__ corr,
                        float* __restrict__ attn_s,
                        float* __restrict__ attn_q) {
  __shared__ float tile[HW * HW];   // 40 KB
  __shared__ float asum[HW];
  __shared__ float qsum[HW];
  const int qn  = blockIdx.x;       // q*WAY + n
  const int tid = threadIdx.x;
  const float* src = corr + (size_t)qn * HW * HW;
  for (int i = tid; i < HW * HW; i += blockDim.x) tile[i] = src[i];
  for (int i = tid; i < HW; i += blockDim.x) { asum[i] = 0.f; qsum[i] = 0.f; }
  __syncthreads();

  const float invT = 0.2f;   // 1 / TEMP_ATTN
  if (tid < HW) {
    // ---- cs: softmax over ij for column kl = tid; attn_s[ij] += p
    {
      const int kl = tid;
      float s = 0.f, s2 = 0.f, xmax = -1e30f;
      for (int ij = 0; ij < HW; ++ij) {
        float x = tile[ij * HW + kl];
        s += x; s2 += x * x; xmax = fmaxf(xmax, x);
      }
      float mean = s * (1.0f / HW);
      float var  = fmaxf((s2 - (float)HW * mean * mean) * (1.0f / (HW - 1)), 0.f);
      float scl  = invT / sqrtf(var + 1e-5f);
      float zmax = (xmax - mean) * scl;
      float den = 0.f;
      for (int ij = 0; ij < HW; ++ij)
        den += __expf((tile[ij * HW + kl] - mean) * scl - zmax);
      float rden = 1.0f / den;
      for (int ij = 0; ij < HW; ++ij) {
        float p = __expf((tile[ij * HW + kl] - mean) * scl - zmax) * rden;
        atomicAdd(&asum[ij], p);
      }
    }
    // ---- cq: softmax over kl for row ij = tid; attn_q[kl] += p
    {
      const float* row = &tile[tid * HW];
      float s = 0.f, s2 = 0.f, xmax = -1e30f;
      for (int kl = 0; kl < HW; ++kl) {
        float x = row[kl];
        s += x; s2 += x * x; xmax = fmaxf(xmax, x);
      }
      float mean = s * (1.0f / HW);
      float var  = fmaxf((s2 - (float)HW * mean * mean) * (1.0f / (HW - 1)), 0.f);
      float scl  = invT / sqrtf(var + 1e-5f);
      float zmax = (xmax - mean) * scl;
      float den = 0.f;
      for (int kl = 0; kl < HW; ++kl)
        den += __expf((row[kl] - mean) * scl - zmax);
      float rden = 1.0f / den;
      for (int kl = 0; kl < HW; ++kl)
        atomicAdd(&qsum[kl], __expf((row[kl] - mean) * scl - zmax) * rden);
    }
  }
  __syncthreads();
  for (int i = tid; i < HW; i += blockDim.x) {
    attn_s[(size_t)qn * HW + i] = asum[i];
    attn_q[(size_t)qn * HW + i] = qsum[i];
  }
}

// ---------------------------------------------------------------------------
// K4: attention-weighted pooling + cosine similarity + /TEMP
// ---------------------------------------------------------------------------
__global__ void k4_sim(const float* __restrict__ spt_se, const float* __restrict__ qry_se,
                       const float* __restrict__ attn_s, const float* __restrict__ attn_q,
                       float* __restrict__ out) {
  __shared__ float as[HW], aq[HW];
  __shared__ float red[3 * 256];
  const int qn = blockIdx.x;
  const int q = qn / WAY, n = qn - q * WAY;
  const int tid = threadIdx.x;
  for (int i = tid; i < HW; i += blockDim.x) {
    as[i] = attn_s[(size_t)qn * HW + i];
    aq[i] = attn_q[(size_t)qn * HW + i];
  }
  __syncthreads();

  float num = 0.f, ns = 0.f, nq = 0.f;
  for (int c = tid; c < CCH; c += blockDim.x) {
    const float* sr = spt_se + ((size_t)n * CCH + c) * HW;
    const float* qr = qry_se + ((size_t)q * CCH + c) * HW;
    float spc = 0.f, qpc = 0.f;
    for (int p = 0; p < HW; ++p) { spc += as[p] * sr[p]; qpc += aq[p] * qr[p]; }
    spc *= (1.0f / HW); qpc *= (1.0f / HW);
    num += spc * qpc; ns += spc * spc; nq += qpc * qpc;
  }
  red[tid] = num; red[256 + tid] = ns; red[512 + tid] = nq;
  __syncthreads();
  for (int st = 128; st > 0; st >>= 1) {
    if (tid < st) {
      red[tid]       += red[tid + st];
      red[256 + tid] += red[256 + tid + st];
      red[512 + tid] += red[512 + tid + st];
    }
    __syncthreads();
  }
  if (tid == 0) {
    float denom = fmaxf(sqrtf(red[256]), 1e-8f) * fmaxf(sqrtf(red[512]), 1e-8f);
    out[qn] = red[0] / denom * 5.0f;   // divide by TEMP = 0.2
  }
}

// ---------------------------------------------------------------------------
// Host launch
// ---------------------------------------------------------------------------
extern "C" void kernel_launch(void* const* d_in, const int* in_sizes, int n_in,
                              void* d_out, int out_size, void* d_ws, size_t ws_size,
                              hipStream_t stream) {
  const float* spt = (const float*)d_in[0];
  const float* qry = (const float*)d_in[1];
  const float* w1  = (const float*)d_in[2];
  const float* b1  = (const float*)d_in[3];
  const float* w2  = (const float*)d_in[4];
  const float* b2  = (const float*)d_in[5];
  float* out = (float*)d_out;

  char* ws = (char*)d_ws;
  float*    spt_se = (float*)(ws + SPT_SE_OFF);
  float*    qry_se = (float*)(ws + QRY_SE_OFF);
  float*    corr   = (float*)(ws + CORR_OFF);
  float*    attn_s = (float*)(ws + ATTN_S_OFF);
  float*    attn_q = (float*)(ws + ATTN_Q_OFF);
  _Float16* sn16   = (_Float16*)(ws + SN16_OFF);
  _Float16* qn16   = (_Float16*)(ws + QN16_OFF);

  // zero f16 operand buffers (pad rows/cols must be zero for unguarded WMMA loads)
  k0_zero<<<512, 256, 0, stream>>>((unsigned int*)(ws + SN16_OFF), F16_BYTES / 4);

  // preprocess: 5 support + 75 query samples
  k1_pre<<<WAY + NQ, 256, 0, stream>>>(spt, qry, w1, b1, w2, b2,
                                       spt_se, qry_se, sn16, qn16);

  // WMMA GEMM: 7 col-tiles x 32 row-tiles x 75 queries, one wave per tile
  k2_corr_wmma<<<dim3(COLS_PAD / 16, ROWS_PAD / 16, NQ), 32, 0, stream>>>(
      sn16, qn16, corr);

  // dual gaussian-normalized softmax attention
  k3_attn<<<NQ * WAY, 128, 0, stream>>>(corr, attn_s, attn_q);

  // pooled cosine similarity
  k4_sim<<<NQ * WAY, 256, 0, stream>>>(spt_se, qry_se, attn_s, attn_q, out);
}
